// ScriptGreedyDecoder_46205258170692
// MI455X (gfx1250) — matrix operand order repaired and checked by reading, not compiled, not run on hardware
//
#include <hip/hip_runtime.h>

#define B_      32
#define T_      48
#define ENC_    1024
#define PRED_   320
#define G4_     1280
#define JH_     512
#define VOC_    29
#define BLANK_  28
#define MAXSYM_ 6
#define NSTEP_  288
#define HSTR_   328   // padded LDS row stride (elems): 656B rows -> conflict-free b128 frags

typedef __bf16 bf16_t;
typedef bf16_t v16bf __attribute__((ext_vector_type(16)));
typedef float  v8f   __attribute__((ext_vector_type(8)));
typedef unsigned int v4u __attribute__((ext_vector_type(4)));

struct Ptrs {
  const float* x; const int* out_lens; const float* emb;
  const float *Wi0, *Wh0, *b0, *Wi1, *Wh1, *b1, *Wj1, *bj1, *Wj2, *bj2;
  unsigned short *xbf, *wj1f, *wj1g, *wi0r, *wh0r, *wi1r, *wh1r, *embbf;
  float *b0r, *b1r, *fpre;
  float *c0p, *c1p, *c0n, *c1n;
  int *lt;
  float *outp;
};

__device__ __forceinline__ unsigned short f2bf(float f) {
  unsigned int u = __float_as_uint(f);
  u += 0x7FFFu + ((u >> 16) & 1u);          // round-to-nearest-even
  return (unsigned short)(u >> 16);
}
__device__ __forceinline__ float sigf(float x) { return 1.0f / (1.0f + expf(-x)); }

union ABu { v16bf v; v4u q[2]; };

// A fragment (16x32 bf16, row-major src[M][K], ld elems):
// lane<16: elems 0..7 = K kb+0..7, elems 8..15 = K kb+16..23 ; lane>=16: +8
__device__ __forceinline__ v16bf load_a_frag(const unsigned short* s, int ld,
                                             int mbase, int kbase, int lane) {
  const unsigned short* p = s + (size_t)(mbase + (lane & 15)) * ld + kbase + ((lane & 16) ? 8 : 0);
  ABu u;
  u.q[0] = *(const v4u*)p;
  u.q[1] = *(const v4u*)(p + 16);
  return u.v;
}

// B fragment (32x16 bf16) from W stored row-major by N: src[N][K], ld elems.
// lane<16: elems 0..7 = K kb+0..7, elems 8..15 = K kb+16..23 ; lane>=16: +8. col N = nb + lane%16.
__device__ __forceinline__ v16bf load_b_frag(const unsigned short* s, int ld,
                                             int nbase, int kbase, int lane) {
  const unsigned short* p = s + (size_t)(nbase + (lane & 15)) * ld + kbase + ((lane & 16) ? 8 : 0);
  ABu u;
  u.q[0] = *(const v4u*)p;
  u.q[1] = *(const v4u*)(p + 16);
  return u.v;
}

__device__ __forceinline__ v8f wmma_bf16(v16bf a, v16bf b, v8f c) {
  return __builtin_amdgcn_wmma_f32_16x16x32_bf16(false, a, false, b, (short)0, c, false, false);
}

// ---------------- phase 0: convert / reorder / init ----------------
__global__ void k_prep(Ptrs p) {
  long tid = (long)blockIdx.x * blockDim.x + threadIdx.x;
  long stride = (long)gridDim.x * blockDim.x;
  for (long i = tid; i < (long)B_ * T_ * ENC_; i += stride) p.xbf[i] = f2bf(p.x[i]);
  for (long i = tid; i < (long)JH_ * ENC_; i += stride) {
    long n = i / ENC_, k = i % ENC_;
    p.wj1f[i] = f2bf(p.Wj1[n * (ENC_ + PRED_) + k]);
  }
  for (long i = tid; i < (long)JH_ * PRED_; i += stride) {
    long n = i / PRED_, k = i % PRED_;
    p.wj1g[i] = f2bf(p.Wj1[n * (ENC_ + PRED_) + ENC_ + k]);
  }
  // gate-interleave reorder: dst row n' = 4*u + g  <-  src row g*PRED + u
  for (long i = tid; i < (long)G4_ * PRED_; i += stride) {
    long np = i / PRED_, k = i % PRED_;
    long u = np >> 2, g = np & 3;
    long sr = (g * PRED_ + u) * PRED_ + k;
    p.wi0r[i] = f2bf(p.Wi0[sr]); p.wh0r[i] = f2bf(p.Wh0[sr]);
    p.wi1r[i] = f2bf(p.Wi1[sr]); p.wh1r[i] = f2bf(p.Wh1[sr]);
  }
  for (long i = tid; i < G4_; i += stride) {
    long u = i >> 2, g = i & 3;
    p.b0r[i] = p.b0[g * PRED_ + u];
    p.b1r[i] = p.b1[g * PRED_ + u];
  }
  for (long i = tid; i < (long)(VOC_ - 1) * PRED_; i += stride) p.embbf[i] = f2bf(p.emb[i]);
  for (long i = tid; i < (long)B_ * PRED_; i += stride) { p.c0p[i] = 0.f; p.c1p[i] = 0.f; }
  for (long i = tid; i < (long)B_ * NSTEP_; i += stride) p.lt[i] = -1;
}

// ---------------- phase 1: Fpre = x @ Wj1f^T + bj1  (M=1536,N=512,K=1024) ----------------
__global__ void k_fpre(Ptrs p) {
  int wave = threadIdx.x >> 5, lane = threadIdx.x & 31;
  int tile = blockIdx.x * (blockDim.x >> 5) + wave;   // 96*32 = 3072 tiles
  if (tile >= 96 * 32) return;
  int mb = (tile >> 5) * 16, nb = (tile & 31) * 16;
  v8f acc = {};
  for (int kk = 0; kk < ENC_; kk += 32)
    acc = wmma_bf16(load_a_frag(p.xbf, ENC_, mb, kk, lane),
                    load_b_frag(p.wj1f, ENC_, nb, kk, lane), acc);
  int n = nb + (lane & 15);
  float bj = p.bj1[n];
  int mofs = (lane & 16) ? 8 : 0;
#pragma unroll
  for (int v = 0; v < 8; v++) {
    int M = mb + v + mofs;
    p.fpre[(size_t)M * JH_ + n] = acc[v] + bj;
  }
}

// ---------------- LSTM layer: A operands in LDS, weights streamed from L2 ----------------
__device__ __forceinline__ void lstm_layer(const unsigned short* A1, const unsigned short* Hprev,
                                           const unsigned short* Wi, const unsigned short* Wh,
                                           const float* bias, const float* Cprev,
                                           unsigned short* Hnew, float* Cnew,
                                           int wave, int lane) {
  for (int tile = wave; tile < 160; tile += 32) {     // 2 Mtiles x 80 Ntiles
    int mb = (tile & 1) * 16, nb = (tile >> 1) * 16;
    v8f acc = {};
    for (int kk = 0; kk < PRED_; kk += 32)
      acc = wmma_bf16(load_a_frag(A1, HSTR_, mb, kk, lane),
                      load_b_frag(Wi, PRED_, nb, kk, lane), acc);
    for (int kk = 0; kk < PRED_; kk += 32)
      acc = wmma_bf16(load_a_frag(Hprev, HSTR_, mb, kk, lane),
                      load_b_frag(Wh, PRED_, nb, kk, lane), acc);
    float bb = bias[nb + (lane & 15)];
    int q = lane & 28;                                 // quad base (keeps half-group bit 4)
    int mofs = (lane & 16) ? 8 : 0;
    int U = (nb >> 2) + ((lane & 12) >> 2);            // unit this lane's quad owns
#pragma unroll
    for (int v = 0; v < 8; v++) {
      float gv = acc[v] + bb;                          // this lane's gate value (incl. bias)
      float iv = __shfl(gv, q + 0, 32);
      float fv = __shfl(gv, q + 1, 32);
      float gg = __shfl(gv, q + 2, 32);
      float ov = __shfl(gv, q + 3, 32);
      if ((lane & 3) == 0) {
        int M = mb + v + mofs;
        float cp = Cprev[M * PRED_ + U];
        float c2 = sigf(fv) * cp + sigf(iv) * tanhf(gg);
        float h2 = sigf(ov) * tanhf(c2);
        Cnew[M * PRED_ + U] = c2;
        Hnew[M * HSTR_ + U] = f2bf(h2);
      }
    }
  }
}

// ---------------- phase 2: persistent sequential decoder (1 WG, 32 waves) ----------------
#define SM_E    0
#define SM_H0P  (SM_E   + B_ * HSTR_)
#define SM_H0N  (SM_H0P + B_ * HSTR_)
#define SM_H1P  (SM_H0N + B_ * HSTR_)
#define SM_H1N  (SM_H1P + B_ * HSTR_)
#define SM_END  (SM_H1N + B_ * HSTR_)          // elems (ushort)
#define SM_RB   ((SM_END * 2 + 15) & ~15)      // byte offset of float rbuf
#define SMEM_BYTES (SM_RB + B_ * JH_ * 4)

__launch_bounds__(1024)
__global__ void k_decode(Ptrs p) {
  extern __shared__ char smem[];
  unsigned short* sh   = (unsigned short*)smem;
  unsigned short* sh_e   = sh + SM_E;
  unsigned short* sh_h0p = sh + SM_H0P;
  unsigned short* sh_h0n = sh + SM_H0N;
  unsigned short* sh_h1p = sh + SM_H1P;
  unsigned short* sh_h1n = sh + SM_H1N;
  float* sh_r = (float*)(smem + SM_RB);

  __shared__ int s_label[B_], s_tsel[B_], s_nb[B_], s_lc[B_], s_ti[B_], s_sa[B_];
  __shared__ float s_logits[B_][VOC_];
  const int tid = threadIdx.x;
  const int wave = tid >> 5, lane = tid & 31;

  if (tid < B_) { s_label[tid] = -1; s_tsel[tid] = 0; s_nb[tid] = 0;
                  s_lc[tid] = 0; s_ti[tid] = 0; s_sa[tid] = 0; }
  for (int i = tid; i < B_ * HSTR_; i += 1024) { sh_h0p[i] = 0; sh_h1p[i] = 0; }
  __syncthreads();

  for (int step = 0; step < NSTEP_; ++step) {
    // ---- embedding into LDS: e = label>=0 ? emb[label] : 0
    for (int i = tid; i < B_ * PRED_; i += 1024) {
      int b = i / PRED_, k = i % PRED_;
      int lab = s_label[b];
      sh_e[b * HSTR_ + k] = (lab >= 0) ? p.embbf[lab * PRED_ + k] : (unsigned short)0;
    }
    __syncthreads();

    // ---- LSTM layer 0 then layer 1 (h handoff via LDS)
    lstm_layer(sh_e,   sh_h0p, p.wi0r, p.wh0r, p.b0r, p.c0p, sh_h0n, p.c0n, wave, lane);
    __syncthreads();
    lstm_layer(sh_h0n, sh_h1p, p.wi1r, p.wh1r, p.b1r, p.c1p, sh_h1n, p.c1n, wave, lane);
    __syncthreads();

    // ---- joint hidden: r = relu(Fpre[b][tsel] + g @ Wj1g^T) into LDS
    for (int tile = wave; tile < 64; tile += 32) {     // 2 Mtiles x 32 Ntiles
      int mb = (tile & 1) * 16, nb = (tile >> 1) * 16;
      v8f acc = {};
      for (int kk = 0; kk < PRED_; kk += 32)
        acc = wmma_bf16(load_a_frag(sh_h1n, HSTR_, mb, kk, lane),
                        load_b_frag(p.wj1g, PRED_, nb, kk, lane), acc);
      int n = nb + (lane & 15);
      int mofs = (lane & 16) ? 8 : 0;
#pragma unroll
      for (int v = 0; v < 8; v++) {
        int M = mb + v + mofs;
        float val = acc[v] + p.fpre[((size_t)M * T_ + s_tsel[M]) * JH_ + n];
        sh_r[M * JH_ + n] = fmaxf(val, 0.f);
      }
    }
    __syncthreads();

    // ---- logits = r @ Wj2^T + bj2  (32x29, K=512, VALU; r from LDS)
    for (int i = tid; i < B_ * VOC_; i += 1024) {
      int b = i / VOC_, v = i % VOC_;
      const float* rr = sh_r + b * JH_;
      const float* w  = p.Wj2 + v * JH_;
      float s = p.bj2[v];
#pragma unroll 4
      for (int k = 0; k < JH_; k++) s += rr[k] * w[k];
      s_logits[b][v] = s;
    }
    __syncthreads();

    // ---- per-batch greedy state update
    if (tid < B_) {
      int b = tid;
      float best = s_logits[b][0]; int kmax = 0;
      for (int v = 1; v < VOC_; v++) {
        float xv = s_logits[b][v];
        if (xv > best) { best = xv; kmax = v; }
      }
      bool blk = (kmax == BLANK_);
      int ti = s_ti[b] + (blk ? 1 : 0);
      bool blank_vec = (ti >= p.out_lens[b]);
      int sa = blk ? 0 : s_sa[b];
      bool nb_ = !(blank_vec || blk);
      int lc = s_lc[b];
      if (nb_) {
        lc += 1;
        if (lc < NSTEP_) p.lt[b * NSTEP_ + lc] += (kmax + 1);   // -1 + (k+1) = k
        sa += 1;
      }
      if (sa >= MAXSYM_) { ti += 1; sa = 0; }
      s_ti[b] = ti; s_sa[b] = sa; s_lc[b] = lc;
      s_tsel[b] = (ti < T_ - 1) ? ti : (T_ - 1);
      s_nb[b] = nb_ ? 1 : 0;
      int rd = (lc < NSTEP_) ? lc : (NSTEP_ - 1);
      s_label[b] = p.lt[b * NSTEP_ + rd];
    }
    __threadfence();   // make c0n/c1n (global) visible before cross-thread commit
    __syncthreads();

    // ---- commit h (LDS) / c (global) for rows that emitted a symbol
    for (int i = tid; i < B_ * PRED_; i += 1024) {
      int b = i / PRED_, u = i % PRED_;
      if (s_nb[b]) {
        sh_h0p[b * HSTR_ + u] = sh_h0n[b * HSTR_ + u];
        sh_h1p[b * HSTR_ + u] = sh_h1n[b * HSTR_ + u];
        p.c0p[i] = p.c0n[i]; p.c1p[i] = p.c1n[i];
      }
    }
    __threadfence();   // committed c visible to owner lanes next step
    __syncthreads();
  }

  // ---- epilogue: outputs as float (lt flat, then lc)
  for (int i = tid; i < B_ * NSTEP_; i += 1024) p.outp[i] = (float)p.lt[i];
  if (tid < B_) p.outp[B_ * NSTEP_ + tid] = (float)s_lc[tid];
}

extern "C" void kernel_launch(void* const* d_in, const int* in_sizes, int n_in,
                              void* d_out, int out_size, void* d_ws, size_t ws_size,
                              hipStream_t stream) {
  (void)in_sizes; (void)n_in; (void)out_size; (void)ws_size;
  char* w = (char*)d_ws;
  size_t off = 0;
  auto alloc = [&](size_t bytes) -> char* {
    char* q = w + off;
    off += (bytes + 255) & ~(size_t)255;
    return q;
  };

  Ptrs P;
  P.x        = (const float*)d_in[0];
  P.out_lens = (const int*)  d_in[1];
  P.emb      = (const float*)d_in[2];
  P.Wi0 = (const float*)d_in[3];  P.Wh0 = (const float*)d_in[4];  P.b0 = (const float*)d_in[5];
  P.Wi1 = (const float*)d_in[6];  P.Wh1 = (const float*)d_in[7];  P.b1 = (const float*)d_in[8];
  P.Wj1 = (const float*)d_in[9];  P.bj1 = (const float*)d_in[10];
  P.Wj2 = (const float*)d_in[11]; P.bj2 = (const float*)d_in[12];

  P.xbf   = (unsigned short*)alloc((size_t)B_ * T_ * ENC_ * 2);
  P.wj1f  = (unsigned short*)alloc((size_t)JH_ * ENC_ * 2);
  P.wj1g  = (unsigned short*)alloc((size_t)JH_ * PRED_ * 2);
  P.wi0r  = (unsigned short*)alloc((size_t)G4_ * PRED_ * 2);
  P.wh0r  = (unsigned short*)alloc((size_t)G4_ * PRED_ * 2);
  P.wi1r  = (unsigned short*)alloc((size_t)G4_ * PRED_ * 2);
  P.wh1r  = (unsigned short*)alloc((size_t)G4_ * PRED_ * 2);
  P.embbf = (unsigned short*)alloc((size_t)(VOC_ - 1) * PRED_ * 2);
  P.b0r   = (float*)alloc((size_t)G4_ * 4);
  P.b1r   = (float*)alloc((size_t)G4_ * 4);
  P.fpre  = (float*)alloc((size_t)B_ * T_ * JH_ * 4);
  P.c0p   = (float*)alloc((size_t)B_ * PRED_ * 4);
  P.c1p   = (float*)alloc((size_t)B_ * PRED_ * 4);
  P.c0n   = (float*)alloc((size_t)B_ * PRED_ * 4);
  P.c1n   = (float*)alloc((size_t)B_ * PRED_ * 4);
  P.lt    = (int*)alloc((size_t)B_ * NSTEP_ * 4);
  P.outp  = (float*)d_out;

  k_prep  <<<512, 256, 0, stream>>>(P);
  k_fpre  <<<384, 256, 0, stream>>>(P);                 // 3072 tiles / 8 waves per block
  k_decode<<<1, 1024, SMEM_BYTES, stream>>>(P);
}
